// SinkhornOTLocalBranch_15049565405743
// MI455X (gfx1250) — matrix operand order repaired and checked by reading, not compiled, not run on hardware
//
#include <hip/hip_runtime.h>
#include <hip/hip_bf16.h>
#include <math.h>

// ---------------------------------------------------------------------------
// Problem constants (from reference)
// ---------------------------------------------------------------------------
constexpr int Qn   = 50;
constexpr int WAYn = 5;
constexpr int SHOTn= 5;
constexpr int Mn   = 196;
constexpr int Dn   = 640;
constexpr int KMn  = SHOTn * Mn;        // 980
constexpr int ROWS = Qn * Mn;           // 9800 flattened (q,m) rows
constexpr int NIT  = 5;
constexpr float VAR_FLOOR = 0.01f;

typedef _Float16 v16h __attribute__((ext_vector_type(16)));
typedef _Float16 v8h  __attribute__((ext_vector_type(8)));
typedef float    v8f  __attribute__((ext_vector_type(8)));

__device__ __forceinline__ float softplus_f(float x) {
    // log(1+exp(x)); inputs here are O(1) so the naive form is fine
    return log1pf(__expf(x));
}

// ---------------------------------------------------------------------------
// Kernel 1: per-way dimension weights  w = d * inv / sum(inv),  inv = 1/(var+floor)
// grid = WAYn blocks, 256 threads
// ---------------------------------------------------------------------------
__global__ void prep_weights_kernel(const float* __restrict__ var,
                                    float* __restrict__ wbuf) {
    __shared__ float sr[256];
    const int w   = blockIdx.x;
    const int tid = threadIdx.x;
    float part = 0.f;
    for (int d = tid; d < Dn; d += 256)
        part += 1.0f / (var[w * Dn + d] + VAR_FLOOR);
    sr[tid] = part;
    __syncthreads();
    for (int s = 128; s > 0; s >>= 1) {
        if (tid < s) sr[tid] += sr[tid + s];
        __syncthreads();
    }
    const float scale = (float)Dn / fmaxf(sr[0], 1e-6f);
    for (int d = tid; d < Dn; d += 256)
        wbuf[w * Dn + d] = (1.0f / (var[w * Dn + d] + VAR_FLOOR)) * scale;
}

// ---------------------------------------------------------------------------
// Kernel 2: support -> f16 with weights folded in:  s16[w][k][d] = f16(S * w[w][d])
// ---------------------------------------------------------------------------
__global__ void conv_support_kernel(const float* __restrict__ sup,
                                    const float* __restrict__ wbuf,
                                    _Float16* __restrict__ s16) {
    const long long total = (long long)WAYn * KMn * Dn;
    for (long long i = (long long)blockIdx.x * blockDim.x + threadIdx.x;
         i < total; i += (long long)gridDim.x * blockDim.x) {
        int w = (int)(i / ((long long)KMn * Dn));
        int d = (int)(i % Dn);
        s16[i] = (_Float16)(sup[i] * wbuf[w * Dn + d]);
    }
}

// ---------------------------------------------------------------------------
// Kernel 3: query tokens -> f16
// ---------------------------------------------------------------------------
__global__ void conv_query_kernel(const float* __restrict__ qt,
                                  _Float16* __restrict__ q16) {
    const long long total = (long long)ROWS * Dn;
    for (long long i = (long long)blockIdx.x * blockDim.x + threadIdx.x;
         i < total; i += (long long)gridDim.x * blockDim.x) {
        q16[i] = (_Float16)qt[i];
    }
}

// ---------------------------------------------------------------------------
// Kernel 4: WMMA GEMM.  One wave computes one 16x16 tile of
//   log_K[q][w][m][k] = (sim - 1) / eps,   sim = Q[row,:] . Sw[k,:]
// grid = (ceil(ROWS/16), ceil(KM/16), WAY), block = 32 (one wave).
// A/B use contiguous k order (element e -> k = half*16 + e), same for A and B.
// ---------------------------------------------------------------------------
__global__ __launch_bounds__(32)
void gemm_logk_kernel(const _Float16* __restrict__ q16,
                      const _Float16* __restrict__ s16,
                      const float* __restrict__ eps_raw,
                      float* __restrict__ logK) {
    const int tileR = blockIdx.x;
    const int tileC = blockIdx.y;
    const int w     = blockIdx.z;

    const int lane = threadIdx.x;       // 0..31, wave32
    const int half = lane >> 4;         // 0 or 1
    const int l15  = lane & 15;

    // A: row = tile row + l15 (same for both lane halves), K-base = half*16
    int arow = tileR * 16 + l15;
    arow = arow < ROWS ? arow : ROWS - 1;          // clamp, keep EXEC all-1s
    const _Float16* Aptr = q16 + (size_t)arow * Dn + half * 16;

    // B: column (support token) = tile col + l15, K-base = half*16
    int bcol = tileC * 16 + l15;
    bcol = bcol < KMn ? bcol : KMn - 1;
    const _Float16* Bptr = s16 + ((size_t)w * KMn + bcol) * Dn + half * 16;

    v8f acc = {};
    for (int k0 = 0; k0 < Dn; k0 += 32) {
        // A element e -> K = k0 + half*16 + e
        v8h alo = *(const v8h*)(Aptr + k0);
        v8h ahi = *(const v8h*)(Aptr + k0 + 8);
        v16h a;
#pragma unroll
        for (int i = 0; i < 8; ++i) { a[i] = alo[i]; a[i + 8] = ahi[i]; }
        // B element e -> K = k0 + half*16 + e  (contiguous 32B)
        v16h b = *(const v16h*)(Bptr + k0);

        acc = __builtin_amdgcn_wmma_f32_16x16x32_f16(
                  false, a, false, b, (short)0, acc, false, false);
    }

    const float eps     = softplus_f(eps_raw[0]) + 1e-4f;
    const float inv_eps = 1.0f / eps;

    // D layout: lane holds column N = l15; VGPR r holds row r + half*8
    const int col = tileC * 16 + l15;
    if (col < KMn) {
#pragma unroll
        for (int r = 0; r < 8; ++r) {
            const int rg = tileR * 16 + r + half * 8;
            if (rg < ROWS) {
                const int q = rg / Mn;
                const int m = rg - q * Mn;
                const size_t idx = (((size_t)q * WAYn + w) * Mn + m) * (size_t)KMn + col;
                logK[idx] = (acc[r] - 1.0f) * inv_eps;
            }
        }
    }
}

// ---------------------------------------------------------------------------
// Kernel 5: Sinkhorn iterations + OT cost contraction.
// One 1024-thread block per (q, way) pair. u[196], v[980] in LDS;
// log_K streamed from workspace (should be L2-resident: 192MB == MI455X L2).
// ---------------------------------------------------------------------------
__global__ __launch_bounds__(1024)
void sinkhorn_kernel(const float* __restrict__ logK_all,
                     const float* __restrict__ eps_raw,
                     const float* __restrict__ tau_raw,
                     float* __restrict__ out) {
    __shared__ float u[Mn];
    __shared__ float v[KMn];
    __shared__ float red[32];

    const int pair = blockIdx.x;                 // pair = q*WAY + w  (matches layout)
    const float* __restrict__ LK = logK_all + (size_t)pair * Mn * KMn;

    const int tid  = threadIdx.x;
    const int wv   = tid >> 5;
    const int ln   = tid & 31;

    const float eps = softplus_f(eps_raw[0]) + 1e-4f;
    const float tau = softplus_f(tau_raw[0]) + 0.01f;
    const float LOG_MU = -logf((float)Mn);
    const float LOG_NU = -logf((float)KMn);

    if (tid < Mn) u[tid] = 0.0f;
    __syncthreads();

    for (int it = 0; it < NIT; ++it) {
        // ---- v update: one thread per column k, online logsumexp over m ----
        if (tid < KMn) {
            float mx = -INFINITY, s = 0.0f;
            for (int m = 0; m < Mn; ++m) {
                const float x = LK[(size_t)m * KMn + tid] + u[m];
                if (x > mx) { s = s * __expf(mx - x) + 1.0f; mx = x; }
                else        { s += __expf(x - mx); }
            }
            v[tid] = LOG_NU - (mx + __logf(s));
        }
        __syncthreads();

        // ---- u update: one wave per row m, lanes stride over k ----
        for (int m = wv; m < Mn; m += 32) {
            float mx = -INFINITY, s = 0.0f;
            const float* row = LK + (size_t)m * KMn;
            for (int k = ln; k < KMn; k += 32) {
                const float x = row[k] + v[k];
                if (x > mx) { s = s * __expf(mx - x) + 1.0f; mx = x; }
                else        { s += __expf(x - mx); }
            }
            // wave32 combine of (max, sum) pairs
            for (int off = 16; off > 0; off >>= 1) {
                const float omx = __shfl_xor(mx, off);
                const float os  = __shfl_xor(s,  off);
                const float nm  = fmaxf(mx, omx);
                s  = s * __expf(mx - nm) + os * __expf(omx - nm);
                mx = nm;
            }
            if (ln == 0) u[m] = LOG_MU - (mx + __logf(s));
        }
        __syncthreads();
    }

    // ---- ot_cost = sum gamma * cost, gamma = exp(logK+u+v), cost = -eps*logK
    float acc = 0.0f;
    for (int m = 0; m < Mn; ++m) {
        const float um = u[m];
        const float* row = LK + (size_t)m * KMn;
        for (int k = tid; k < KMn; k += 1024) {
            const float lk = row[k];
            acc += __expf(lk + um + v[k]) * (-eps * lk);
        }
    }
    for (int off = 16; off > 0; off >>= 1) acc += __shfl_xor(acc, off);
    if (ln == 0) red[wv] = acc;
    __syncthreads();
    if (wv == 0) {
        float a2 = red[ln];
        for (int off = 16; off > 0; off >>= 1) a2 += __shfl_xor(a2, off);
        if (ln == 0) out[pair] = -a2 / tau;
    }
}

// ---------------------------------------------------------------------------
// Launcher
// ---------------------------------------------------------------------------
extern "C" void kernel_launch(void* const* d_in, const int* in_sizes, int n_in,
                              void* d_out, int out_size, void* d_ws, size_t ws_size,
                              hipStream_t stream) {
    const float* qtok    = (const float*)d_in[0]; // [Q, M, D]
    const float* sup     = (const float*)d_in[1]; // [WAY, SHOT, M, D] == [WAY, KM, D]
    const float* var     = (const float*)d_in[2]; // [WAY, D]
    const float* eps_raw = (const float*)d_in[3]; // scalar
    const float* tau_raw = (const float*)d_in[4]; // scalar
    float* out = (float*)d_out;                   // [Q, WAY]

    // Workspace layout (all 16B-aligned):
    char* ws = (char*)d_ws;
    size_t off = 0;
    float* logK = (float*)(ws + off);             // 50*5*196*980*4 = 192,080,000 B
    off += (size_t)Qn * WAYn * Mn * KMn * sizeof(float);
    _Float16* q16 = (_Float16*)(ws + off);        // 9800*640*2 = 12,544,000 B
    off += (size_t)ROWS * Dn * sizeof(_Float16);
    _Float16* s16 = (_Float16*)(ws + off);        // 5*980*640*2 = 6,272,000 B
    off += (size_t)WAYn * KMn * Dn * sizeof(_Float16);
    float* wbuf = (float*)(ws + off);             // 5*640*4
    off += (size_t)WAYn * Dn * sizeof(float);

    // 1) dimension weights per way
    prep_weights_kernel<<<WAYn, 256, 0, stream>>>(var, wbuf);

    // 2) support -> f16 with weights folded in
    {
        const long long total = (long long)WAYn * KMn * Dn;
        int blocks = (int)((total + 255) / 256);
        if (blocks > 4096) blocks = 4096;
        conv_support_kernel<<<blocks, 256, 0, stream>>>(sup, wbuf, s16);
    }

    // 3) query -> f16
    {
        const long long total = (long long)ROWS * Dn;
        int blocks = (int)((total + 255) / 256);
        if (blocks > 4096) blocks = 4096;
        conv_query_kernel<<<blocks, 256, 0, stream>>>(qtok, q16);
    }

    // 4) WMMA GEMM -> log_K
    {
        dim3 grid((ROWS + 15) / 16, (KMn + 15) / 16, WAYn); // 613 x 62 x 5
        gemm_logk_kernel<<<grid, 32, 0, stream>>>(q16, s16, eps_raw, logK);
    }

    // 5) Sinkhorn + contraction -> out
    sinkhorn_kernel<<<Qn * WAYn, 1024, 0, stream>>>(logK, eps_raw, tau_raw, out);
}
